// Detector_61984968016074
// MI455X (gfx1250) — matrix operand order repaired and checked by reading, not compiled, not run on hardware
//
#include <hip/hip_runtime.h>
#include <hip/hip_bf16.h>
#include <float.h>

// ---------------------------------------------------------------------------
// Problem constants (from the reference)
// ---------------------------------------------------------------------------
#define C_CH   256
#define H_FT   50
#define W_FT   50
#define S_BIN  7
#define KH_B   9          // H//S + 2
#define KW_B   9          // W//S + 2
#define NB     128        // boxes (M dimension of the GEMMs)
#define DFEAT  12544      // C*S*S  (K of fc1)
#define NHID   4096
#define NCLS   21
#define NREG   84

typedef __attribute__((ext_vector_type(16))) __bf16 v16bf;
typedef __attribute__((ext_vector_type(8)))  float  v8f;

// ---------------------------------------------------------------------------
// 2 x f32 -> packed bf16.
//   1) hardware packed converter if the toolchain declares it (1 VALU, RNE)
//   2) v_perm_b32 byte-select (1 VALU, truncation)
//   3) RNE bit-twiddle fallback
// ---------------------------------------------------------------------------
static __device__ __forceinline__ unsigned pk_bf16(float a, float b) {
#if __has_builtin(__builtin_amdgcn_cvt_pk_bf16_f32)
    typedef __attribute__((ext_vector_type(2))) __bf16 v2bf;
    v2bf p = __builtin_amdgcn_cvt_pk_bf16_f32(a, b);
    return __builtin_bit_cast(unsigned, p);
#elif __has_builtin(__builtin_amdgcn_perm)
    // result bytes {b3,b2,a3,a2}: hi16(b) | hi16(a) -> packed {bf16(b),bf16(a)}
    return __builtin_amdgcn_perm(__builtin_bit_cast(unsigned, b),
                                 __builtin_bit_cast(unsigned, a),
                                 0x07060302u);
#else
    unsigned ua = __builtin_bit_cast(unsigned, a);
    unsigned ub = __builtin_bit_cast(unsigned, b);
    unsigned lo = (ua + 0x7FFFu + ((ua >> 16) & 1u)) >> 16;
    unsigned hi = (ub + 0x7FFFu + ((ub >> 16) & 1u)) >> 16;
    return (lo & 0xFFFFu) | (hi << 16);
#endif
}
static __device__ __forceinline__ unsigned short f2bf(float x) {
    return (unsigned short)(pk_bf16(x, 0.0f) & 0xFFFFu);
}
static __device__ __forceinline__ float bf2f(unsigned short h) {
    return __builtin_bit_cast(float, ((unsigned)h) << 16);
}

// ---------------------------------------------------------------------------
// Kernel 1: ROI max-pool, writes A matrix [NB, DFEAT] as bf16 (row-major,
// K-contiguous) — the layout both WMMA GEMMs consume.
// ---------------------------------------------------------------------------
__global__ __launch_bounds__(256) void roi_pool_kernel(
    const float* __restrict__ feat,    // [C,H,W]
    const float* __restrict__ boxes,   // [NB,4]
    unsigned short* __restrict__ A)    // [NB, DFEAT] bf16
{
    int idx = blockIdx.x * 256 + threadIdx.x;
    if (idx >= NB * C_CH * S_BIN * S_BIN) return;
    int j = idx % S_BIN;
    int i = (idx / S_BIN) % S_BIN;
    int c = (idx / (S_BIN * S_BIN)) % C_CH;
    int n = idx / (S_BIN * S_BIN * C_CH);

    const float inv_scale = 1.0f / 16.0f;
    int x0 = (int)(boxes[n * 4 + 0] * inv_scale);
    int y0 = (int)(boxes[n * 4 + 1] * inv_scale);
    int x1 = (int)(boxes[n * 4 + 2] * inv_scale);
    int y1 = (int)(boxes[n * 4 + 3] * inv_scale);
    int lh = max(y1 - y0 + 1, 1);
    int lw = max(x1 - x0 + 1, 1);

    int ys = y0 + (i * lh) / S_BIN;
    int ye = y0 + ((i + 1) * lh + S_BIN - 1) / S_BIN;   // ceil
    int xs = x0 + (j * lw) / S_BIN;
    int xe = x0 + ((j + 1) * lw + S_BIN - 1) / S_BIN;   // ceil

    const float* fc = feat + (size_t)c * (H_FT * W_FT);
    float m = -FLT_MAX;
    #pragma unroll 1
    for (int ky = 0; ky < KH_B; ++ky) {
        int yi = ys + ky;
        if (yi >= ye) break;
        int y = min(max(yi, 0), H_FT - 1);
        const float* fr = fc + y * W_FT;
        #pragma unroll 1
        for (int kx = 0; kx < KW_B; ++kx) {
            int xi = xs + kx;
            if (xi >= xe) break;
            int x = min(max(xi, 0), W_FT - 1);
            float v = fr[x];
            m = v > m ? v : m;
        }
    }
    A[(size_t)n * DFEAT + c * (S_BIN * S_BIN) + i * S_BIN + j] = f2bf(m);
}

// ---------------------------------------------------------------------------
// Kernel 2: Y[128,N] = relu(A[128,K](bf16) * W[N,K](f32)^T + bias), Y bf16.
//
// Block = 256 threads = 8 waves.  Block tile: M=128 x N=64, K-step 32.
// Weight-bandwidth-bound (AI = 2*128/4B = 64 FLOP/B < bf16 machine balance),
// so the f32 weight tile is streamed from HBM exactly once per element,
// converted to packed bf16 in one v_perm/v_cvt per pair during cooperative
// staging, and consumed from LDS by pure ds_load_b128 fragment builds.
// All 4 B fragments are preloaded so the 4 WMMAs issue back-to-back.
// ---------------------------------------------------------------------------
__global__ __launch_bounds__(256) void gemm_bias_relu_kernel(
    const unsigned short* __restrict__ Abf,  // [128, K] bf16
    const float* __restrict__ W,             // [N, K] f32
    const float* __restrict__ bias,          // [N]
    unsigned short* __restrict__ Y,          // [128, N] bf16
    int K, int N)
{
    __shared__ unsigned ldsB[64 * 16];   // 64 rows x 32 bf16 (16 packed pairs)

    const int tid   = threadIdx.x;
    const int lane  = tid & 31;
    const int wave  = tid >> 5;
    const int lhalf = (lane < 16) ? 0 : 1;   // lane half selects K sub-range
    const int lpos  = lane & 15;
    const int n0    = blockIdx.x * 64;
    const int m0    = wave * 16;

    v8f acc[4];
    #pragma unroll
    for (int t = 0; t < 4; ++t) acc[t] = (v8f){0.f,0.f,0.f,0.f,0.f,0.f,0.f,0.f};

    // A row for this lane, viewed as packed bf16 pairs (uint)
    const unsigned* Arow = reinterpret_cast<const unsigned*>(Abf)
                         + (size_t)(m0 + lpos) * (unsigned)(K >> 1);

    // staging coordinates for this thread (2 float4 per K-step)
    const int srow0 = (tid * 2) >> 3;          // row of first float4
    const int scol0 = ((tid * 2) & 7) << 2;    // f32 col of first float4

    for (int kt = 0; kt < K; kt += 32) {
        // ---- cooperative stage: 64x32 f32 -> packed bf16 in LDS -----------
        {
            #pragma unroll
            for (int q = 0; q < 2; ++q) {
                int f   = tid * 2 + q;
                int row = f >> 3;
                int col = (f & 7) << 2;
                const float* src = &W[(size_t)(n0 + row) * K + kt + col];
                const float4 w4 = *reinterpret_cast<const float4*>(src);
                uint2 p;
                p.x = pk_bf16(w4.x, w4.y);
                p.y = pk_bf16(w4.z, w4.w);
                *reinterpret_cast<uint2*>(&ldsB[row * 16 + ((f & 7) << 1)]) = p;
                // speculative prefetch of next tile (silently dropped at end)
                __builtin_prefetch(src + 32, 0, 1);
            }
        }
        __syncthreads();

        // ---- A fragment: 16x32 bf16.  Lanes<16 hold K 0-7 (v0-3) and
        // 16-23 (v4-7); lanes>=16 hold K 8-15 / 24-31.  Each half is 16B
        // contiguous in K-major storage -> two b128 loads. ------------------
        union frag_t { unsigned u[8]; v16bf v; };
        frag_t afrag;
        {
            const unsigned* p = Arow + (unsigned)(kt >> 1) + lhalf * 4;
            uint4 a0 = *reinterpret_cast<const uint4*>(p);      // K: koff..koff+7
            uint4 a1 = *reinterpret_cast<const uint4*>(p + 8);  // K: 16+koff..
            afrag.u[0] = a0.x; afrag.u[1] = a0.y; afrag.u[2] = a0.z; afrag.u[3] = a0.w;
            afrag.u[4] = a1.x; afrag.u[5] = a1.y; afrag.u[6] = a1.z; afrag.u[7] = a1.w;
        }

        // ---- preload all 4 B fragments (8 x ds_load_b128 as one clause) ---
        frag_t bfr[4];
        #pragma unroll
        for (int t = 0; t < 4; ++t) {
            const uint4* bp = reinterpret_cast<const uint4*>(
                &ldsB[(t * 16 + lpos) * 16 + lhalf * 4]);
            uint4 b0 = bp[0];
            uint4 b1 = bp[2];
            bfr[t].u[0] = b0.x; bfr[t].u[1] = b0.y; bfr[t].u[2] = b0.z; bfr[t].u[3] = b0.w;
            bfr[t].u[4] = b1.x; bfr[t].u[5] = b1.y; bfr[t].u[6] = b1.z; bfr[t].u[7] = b1.w;
        }

        // ---- 4 back-to-back WMMAs ----------------------------------------
        #pragma unroll
        for (int t = 0; t < 4; ++t) {
            acc[t] = __builtin_amdgcn_wmma_f32_16x16x32_bf16(
                false, afrag.v, false, bfr[t].v,
                (short)0, acc[t], false, false);
        }
        __syncthreads();
    }
    (void)srow0; (void)scol0;

    // ---- epilogue: bias + relu, store bf16 --------------------------------
    #pragma unroll
    for (int t = 0; t < 4; ++t) {
        int nn = n0 + t * 16 + lpos;
        float b = bias[nn];
        #pragma unroll
        for (int r = 0; r < 8; ++r) {
            int m = m0 + r + lhalf * 8;   // C/D layout: lanes>=16 are M rows +8
            float v = acc[t][r] + b;
            v = v > 0.f ? v : 0.f;
            Y[(size_t)m * N + nn] = f2bf(v);
        }
    }
}

// ---------------------------------------------------------------------------
// Kernel 3: tiny cls/reg head.  out order: [128,21] cls then [128,84] reg.
// ---------------------------------------------------------------------------
__global__ __launch_bounds__(256) void head_kernel(
    const unsigned short* __restrict__ X,    // [128, 4096] bf16
    const float* __restrict__ cls_w, const float* __restrict__ cls_b,
    const float* __restrict__ reg_w, const float* __restrict__ reg_b,
    float* __restrict__ out)
{
    int idx = blockIdx.x * 256 + threadIdx.x;
    if (idx >= NB * (NCLS + NREG)) return;
    int o = idx % (NCLS + NREG);
    int n = idx / (NCLS + NREG);

    const float* w;
    float b;
    float* dst;
    if (o < NCLS) { w = cls_w + (size_t)o * NHID;          b = cls_b[o];
                    dst = out + (size_t)n * NCLS + o; }
    else          { int o2 = o - NCLS;
                    w = reg_w + (size_t)o2 * NHID;         b = reg_b[o2];
                    dst = out + (size_t)NB * NCLS + (size_t)n * NREG + o2; }

    const unsigned* xr = reinterpret_cast<const unsigned*>(X + (size_t)n * NHID);
    float acc = b;
    #pragma unroll 4
    for (int k = 0; k < NHID / 2; ++k) {
        unsigned p  = xr[k];
        const float2 w2 = *reinterpret_cast<const float2*>(&w[2 * k]);
        acc += bf2f((unsigned short)(p & 0xFFFFu)) * w2.x;
        acc += bf2f((unsigned short)(p >> 16))     * w2.y;
    }
    *dst = acc;
}

// ---------------------------------------------------------------------------
// Launch
// ---------------------------------------------------------------------------
extern "C" void kernel_launch(void* const* d_in, const int* in_sizes, int n_in,
                              void* d_out, int out_size, void* d_ws, size_t ws_size,
                              hipStream_t stream) {
    const float* feat   = (const float*)d_in[0];
    const float* boxes  = (const float*)d_in[1];
    const float* fc1_w  = (const float*)d_in[2];
    const float* fc1_b  = (const float*)d_in[3];
    const float* fc2_w  = (const float*)d_in[4];
    const float* fc2_b  = (const float*)d_in[5];
    const float* cls_w  = (const float*)d_in[6];
    const float* cls_b  = (const float*)d_in[7];
    const float* reg_w  = (const float*)d_in[8];
    const float* reg_b  = (const float*)d_in[9];
    float* out = (float*)d_out;

    // workspace layout (bytes)
    char* ws = (char*)d_ws;
    unsigned short* A  = (unsigned short*)(ws);                       // 128*12544*2 = 3,211,264 (6,422,528 reserved)
    unsigned short* X1 = (unsigned short*)(ws + 6422528);             // 128*4096*2  = 1,048,576
    unsigned short* X2 = (unsigned short*)(ws + 6422528 + 1048576);   // 128*4096*2  = 1,048,576

    // 1. ROI pool -> bf16 A matrix
    {
        int total = NB * C_CH * S_BIN * S_BIN;
        roi_pool_kernel<<<(total + 255) / 256, 256, 0, stream>>>(feat, boxes, A);
    }
    // 2. fc1: [128,12544] x [4096,12544]^T -> relu -> X1 bf16
    gemm_bias_relu_kernel<<<NHID / 64, 256, 0, stream>>>(A, fc1_w, fc1_b, X1, DFEAT, NHID);
    // 3. fc2: [128,4096] x [4096,4096]^T -> relu -> X2 bf16
    gemm_bias_relu_kernel<<<NHID / 64, 256, 0, stream>>>(X1, fc2_w, fc2_b, X2, NHID, NHID);
    // 4. cls + reg heads -> d_out
    {
        int total = NB * (NCLS + NREG);
        head_kernel<<<(total + 255) / 256, 256, 0, stream>>>(X2, cls_w, cls_b, reg_w, reg_b, out);
    }
}
